// CornerPool_59433757442225
// MI455X (gfx1250) — matrix operand order repaired and checked by reading, compiled, not run on hardware
//
#include <hip/hip_runtime.h>

// ---------------------------------------------------------------------------
// CornerPool pipeline for gfx1250 (MI455X): implicit-GEMM convs on
// v_wmma_f32_16x16x32_bf16 (wave32), NHWC-bf16 internal layout.
// Double-buffered LDS staging via CDNA5 async copies, software-pipelined so
// global->LDS fills overlap the XDL WMMA stream (s_wait_asynccnt G).
// ---------------------------------------------------------------------------

typedef __bf16 bf16_t;
typedef __bf16 v16bf __attribute__((ext_vector_type(16)));
typedef __bf16 v8bf  __attribute__((ext_vector_type(8)));
typedef float  v8f   __attribute__((ext_vector_type(8)));

#define NB   4
#define DH   128
#define DW   128
#define DC   256
#define HWSZ (DH * DW)          // 16384 spatial per image
#define MTOT (NB * HWSZ)        // 65536 GEMM rows

// 16-byte async copy global -> LDS (CDNA5).  lds_off = low 32 bits of the
// generic shared address (== DS byte address per ISA aperture rules).
__device__ __forceinline__ void async_copy16(unsigned lds_off, const void* gsrc) {
    asm volatile("global_load_async_to_lds_b128 %0, %1, off"
                 :: "v"(lds_off), "v"(gsrc)
                 : "memory");
}

// ---------------------------------------------------------------------------
// NCHW f32 -> NHWC bf16, LDS-tiled transpose (32x32 tiles, both sides coalesced)
// ---------------------------------------------------------------------------
__global__ void nchw_to_nhwc_bf16(const float* __restrict__ in, bf16_t* __restrict__ out) {
    __shared__ float tile[32][33];
    const int img = blockIdx.z;
    const int c0  = blockIdx.y * 32;
    const int s0  = blockIdx.x * 32;
    const int tx = threadIdx.x, ty = threadIdx.y;   // 32 x 8
    const float* src = in + (size_t)img * DC * HWSZ;
    #pragma unroll
    for (int j = 0; j < 32; j += 8)
        tile[ty + j][tx] = src[(size_t)(c0 + ty + j) * HWSZ + s0 + tx];
    __syncthreads();
    bf16_t* dst = out + (size_t)img * HWSZ * DC;
    #pragma unroll
    for (int j = 0; j < 32; j += 8)
        dst[(size_t)(s0 + ty + j) * DC + c0 + tx] = (bf16_t)tile[tx][ty + j];
}

// ---------------------------------------------------------------------------
// Pack OIHW f32 weights into WMMA B-fragment order:
//   wf[((tap*KC + kc)*NT + nt)*512 + n16*32 + k]   (bf16)
// ---------------------------------------------------------------------------
__global__ void wprep(const float* __restrict__ w1, const float* __restrict__ w2,
                      int split, int CIN, int COUT, int TAPS, bf16_t* __restrict__ wf) {
    const int total = TAPS * CIN * COUT;
    int t = blockIdx.x * blockDim.x + threadIdx.x;
    if (t >= total) return;
    const int KC = CIN >> 5, NT = COUT >> 4;
    const int k    = t & 31;
    const int n16  = (t >> 5) & 15;
    int rest = t >> 9;
    const int nt  = rest % NT; rest /= NT;
    const int kc  = rest % KC;
    const int tap = rest / KC;
    const int oc = nt * 16 + n16;
    const int ic = kc * 32 + k;
    const float* w = (oc < split) ? w1 : w2;
    const int ocl  = (oc < split) ? oc : oc - split;
    wf[t] = (bf16_t)w[((size_t)ocl * CIN + ic) * TAPS + tap];
}

// ---------------------------------------------------------------------------
// BN folding: scale = g*rsqrt(v+eps), shift = b - m*scale
// ---------------------------------------------------------------------------
__global__ void bnprep(const float* g1, const float* b1, const float* m1, const float* v1,
                       const float* g2, const float* b2, const float* m2, const float* v2,
                       int split, int n, float* __restrict__ sc, float* __restrict__ sh) {
    int i = blockIdx.x * blockDim.x + threadIdx.x;
    if (i >= n) return;
    const int j = (i < split) ? i : i - split;
    const float g = (i < split) ? g1[j] : g2[j];
    const float b = (i < split) ? b1[j] : b2[j];
    const float m = (i < split) ? m1[j] : m2[j];
    const float v = (i < split) ? v1[j] : v2[j];
    const float s = g * rsqrtf(v + 1e-5f);
    sc[i] = s;
    sh[i] = b - m * s;
}

// ---------------------------------------------------------------------------
// Implicit-GEMM conv, 4 waves / block, 64(M) x 64(N) block tile.
// Software-pipelined, double-buffered LDS staging over a FIXED tap count:
//   stage i+1 async-copies (A strip + B panel) are issued before computing
//   stage i; s_wait_asynccnt <= G (G = per-wave async ops per stage, in-order
//   completion) proves stage i landed while stage i+1 streams behind the
//   WMMAs.  Out-of-image rows (h or w) are fetched from clamped addresses so
//   the per-wave async count is uniform, then zero-fixed by the same thread
//   that issued them (writer == fixer: no cross-wave race).
// A-frag (ISA 7.12.2): lane m=lane&15, half=lane>>4; two 16B LDS reads.
// B-frag: lane n=lane&15; two 16B LDS reads at K=half*8 and K=16+half*8.
// LDS padding: A rows +16B (bank rotate x4), B 64B groups at 80B stride.
// OMODE: 0 = f32 NHWC, 1 = bf16 NHWC, 2 = f32 NCHW (final output).
// ---------------------------------------------------------------------------
template <int CIN, int COUT, int TAPS, bool RELU, int OMODE, bool ADDRES>
__global__ __launch_bounds__(128)
void conv_wmma(const bf16_t* __restrict__ in, const bf16_t* __restrict__ wf,
               const float* __restrict__ sc, const float* __restrict__ sh,
               const float* __restrict__ res, void* __restrict__ outv) {
    constexpr int KC      = CIN / 32;           // K chunks of 32
    constexpr int NT      = COUT / 16;
    constexpr int SEGROW  = CIN / 8;            // 16B segments per A row
    constexpr int AROWB   = CIN * 2 + 16;       // padded A row bytes
    constexpr int NSEG_A  = 64 * SEGROW;
    constexpr int BCHUNKB = 4 * 1024 + 1024;    // 4KB panel + 16B pad per 64B group
    constexpr int NSEG_B  = KC * 256;           // 16B segments of B (unpadded)
    constexpr int STAGE   = 64 * AROWB + KC * BCHUNKB;
    constexpr int G       = NSEG_A / 128 + NSEG_B / 128;  // async ops / wave / stage

    extern __shared__ char smem[];              // 2 * STAGE bytes (dynamic LDS)
    const unsigned smoff = (unsigned)(uintptr_t)smem;

    const int tid  = threadIdx.x;
    const int lane = tid & 31;
    const int wv   = tid >> 5;
    const int n16  = lane & 15;
    const int hl   = lane >> 4;
    const int mb   = blockIdx.x * 64;           // 64 consecutive spatial rows
    const int nt0  = blockIdx.y * 4;

    const int img = mb >> 14;
    const int h   = (mb >> 7) & 127;            // uniform over the tile
    const int w0  = mb & 127;                   // 0 or 64

    auto issue = [&](int tap, int p) {
        const int kh = (TAPS == 9) ? (tap / 3 - 1) : 0;
        const int kw = (TAPS == 9) ? (tap % 3 - 1) : 0;
        int hh = h + kh;
        hh = hh < 0 ? 0 : (hh > DH - 1 ? DH - 1 : hh);       // clamp (uniform)
        const bf16_t* grow = in + ((size_t)img * DH + hh) * DW * CIN;
        const unsigned sAo = smoff + p * STAGE;
        const unsigned sBo = sAo + 64 * AROWB;
        for (int s = tid; s < NSEG_A; s += 128) {
            const int row = s / SEGROW;
            const int off = s % SEGROW;
            int ww = w0 + row + kw;
            ww = ww < 0 ? 0 : (ww > DW - 1 ? DW - 1 : ww);   // clamp: uniform count
            async_copy16(sAo + row * AROWB + off * 16,
                         grow + (size_t)ww * CIN + off * 8);
        }
        for (int s = tid; s < NSEG_B; s += 128) {
            const int kc = s >> 8;
            const int j  = s & 255;
            async_copy16(sBo + kc * BCHUNKB + j * 16 + (j >> 2) * 16,
                         wf + (((size_t)tap * KC + kc) * NT + nt0) * 512 + j * 8);
        }
    };

    issue(0, 0);

    v8f acc[4] = {};

    #pragma unroll 1
    for (int tap = 0; tap < TAPS; ++tap) {
        if (tap + 1 < TAPS) {
            issue(tap + 1, (tap + 1) & 1);
            // stage `tap` done once outstanding <= G (in-order async completion)
            asm volatile("s_wait_asynccnt %0" :: "n"(G) : "memory");
        } else {
            asm volatile("s_wait_asynccnt 0x0" ::: "memory");
        }
        const int p = tap & 1;
        char* sA = smem + p * STAGE;
        char* sB = sA + 64 * AROWB;

        // zero-fix clamped out-of-image segments (same thread that issued them)
        if (TAPS == 9) {
            const int kh = tap / 3 - 1;
            const int kw = tap % 3 - 1;
            const bool hbad = (unsigned)(h + kh) >= DH;      // uniform
            if (hbad || kw != 0) {
                for (int s = tid; s < NSEG_A; s += 128) {
                    const int row = s / SEGROW;
                    const int ww  = w0 + row + kw;
                    if (hbad || (unsigned)ww >= DW)
                        *(float4*)(sA + row * AROWB + (s % SEGROW) * 16) =
                            make_float4(0.f, 0.f, 0.f, 0.f);
                }
            }
        }
        __syncthreads();

        // ---- K loop from LDS ----
        const int arow = wv * 16 + n16;
        #pragma unroll
        for (int kc = 0; kc < KC; ++kc) {
            v8bf alo = *(const v8bf*)(sA + arow * AROWB + kc * 64 + hl * 16);
            v8bf ahi = *(const v8bf*)(sA + arow * AROWB + kc * 64 + 32 + hl * 16);
            v16bf a = __builtin_shufflevector(alo, ahi, 0, 1, 2, 3, 4, 5, 6, 7,
                                              8, 9, 10, 11, 12, 13, 14, 15);
            #pragma unroll
            for (int t4 = 0; t4 < 4; ++t4) {
                const char* bp = sB + kc * BCHUNKB + (t4 * 16 + n16) * 80 + hl * 16;
                v8bf blo = *(const v8bf*)(bp);
                v8bf bhi = *(const v8bf*)(bp + 32);
                v16bf b = __builtin_shufflevector(blo, bhi, 0, 1, 2, 3, 4, 5, 6, 7,
                                                  8, 9, 10, 11, 12, 13, 14, 15);
                acc[t4] = __builtin_amdgcn_wmma_f32_16x16x32_bf16(
                    false, a, false, b, (short)0, acc[t4], false, false);
            }
        }
        __syncthreads();    // all ds reads done before this buffer is refilled
    }

    // ---- epilogue: BN (+ residual) (+ ReLU), store ----
    const int mwb = mb + wv * 16;
    #pragma unroll
    for (int s = 0; s < 4; ++s) {
        const int oc = (nt0 + s) * 16 + n16;
        const float scale = sc[oc];
        const float shift = sh[oc];
        #pragma unroll
        for (int v = 0; v < 8; ++v) {
            const int mg = mwb + v + 8 * hl;
            float val = acc[s][v] * scale + shift;
            if (ADDRES) val += res[(size_t)mg * COUT + oc];
            if (RELU) val = fmaxf(val, 0.0f);
            if (OMODE == 0) {
                ((float*)outv)[(size_t)mg * COUT + oc] = val;
            } else if (OMODE == 1) {
                ((bf16_t*)outv)[(size_t)mg * COUT + oc] = (bf16_t)val;
            } else {
                const int gi = mg >> 14, gh = (mg >> 7) & 127, gw = mg & 127;
                ((float*)outv)[(((size_t)gi * COUT + oc) * DH + gh) * DW + gw] = val;
            }
        }
    }
}

// ---------------------------------------------------------------------------
// Reverse cummax over H on channels [0,128) of an NHWC-f32 256-ch buffer.
// ---------------------------------------------------------------------------
__global__ void cummax_h(float* __restrict__ buf) {
    const int t   = blockIdx.x * blockDim.x + threadIdx.x;   // 65536 threads
    const int c   = t & 127;
    const int w   = (t >> 7) & 127;
    const int img = t >> 14;
    float run = -3.402823e38f;
    for (int h = DH - 1; h >= 0; --h) {
        const size_t idx = (((size_t)img * DH + h) * DW + w) * DC + c;
        run = fmaxf(run, buf[idx]);
        buf[idx] = run;
    }
}

// Reverse cummax over W on channels [128,256).
__global__ void cummax_w(float* __restrict__ buf) {
    const int t   = blockIdx.x * blockDim.x + threadIdx.x;   // 65536 threads
    const int c   = 128 + (t & 127);
    const int h   = (t >> 7) & 127;
    const int img = t >> 14;
    float run = -3.402823e38f;
    for (int w = DW - 1; w >= 0; --w) {
        const size_t idx = (((size_t)img * DH + h) * DW + w) * DC + c;
        run = fmaxf(run, buf[idx]);
        buf[idx] = run;
    }
}

// s[m, c] = p1[m, c] + p2[m, c+128]  ->  bf16 NHWC (128 ch)
__global__ void sum_to_bf16(const float* __restrict__ buf, bf16_t* __restrict__ s) {
    const size_t t = (size_t)blockIdx.x * blockDim.x + threadIdx.x;  // 8388608
    const int c    = (int)(t & 127);
    const size_t m = t >> 7;
    s[t] = (bf16_t)(buf[m * DC + c] + buf[m * DC + 128 + c]);
}

// ---------------------------------------------------------------------------
extern "C" void kernel_launch(void* const* d_in, const int* in_sizes, int n_in,
                              void* d_out, int out_size, void* d_ws, size_t ws_size,
                              hipStream_t stream) {
    (void)in_sizes; (void)n_in; (void)out_size; (void)ws_size;

    const float* x    = (const float*)d_in[0];
    const float* w_p1 = (const float*)d_in[1];
    const float* g_p1 = (const float*)d_in[2];
    const float* b_p1 = (const float*)d_in[3];
    const float* m_p1 = (const float*)d_in[4];
    const float* v_p1 = (const float*)d_in[5];
    const float* w_p2 = (const float*)d_in[6];
    const float* g_p2 = (const float*)d_in[7];
    const float* b_p2 = (const float*)d_in[8];
    const float* m_p2 = (const float*)d_in[9];
    const float* v_p2 = (const float*)d_in[10];
    const float* w_p  = (const float*)d_in[11];
    const float* g_p  = (const float*)d_in[12];
    const float* b_p  = (const float*)d_in[13];
    const float* m_p  = (const float*)d_in[14];
    const float* v_p  = (const float*)d_in[15];
    const float* w_c1 = (const float*)d_in[16];
    const float* g_c1 = (const float*)d_in[17];
    const float* b_c1 = (const float*)d_in[18];
    const float* m_c1 = (const float*)d_in[19];
    const float* v_c1 = (const float*)d_in[20];
    const float* w_c2 = (const float*)d_in[21];
    const float* g_c2 = (const float*)d_in[22];
    const float* b_c2 = (const float*)d_in[23];
    const float* m_c2 = (const float*)d_in[24];
    const float* v_c2 = (const float*)d_in[25];

    char* ws = (char*)d_ws;
    size_t off = 0;
    auto alloc = [&](size_t bytes) -> void* {
        void* p = ws + off;
        off += (bytes + 255) & ~(size_t)255;
        return p;
    };
    bf16_t* xh   = (bf16_t*)alloc((size_t)MTOT * DC * 2);        // 33.5 MB
    float*  buf1 = (float*) alloc((size_t)MTOT * DC * 4);        // 67  MB
    bf16_t* sbuf = (bf16_t*)alloc((size_t)MTOT * 128 * 2);       // 16.8 MB
    bf16_t* pbuf = (bf16_t*)alloc((size_t)MTOT * DC * 2);        // 33.5 MB
    bf16_t* wf12 = (bf16_t*)alloc((size_t)9 * 256 * 256 * 2);
    bf16_t* wfp  = (bf16_t*)alloc((size_t)9 * 128 * 256 * 2);
    bf16_t* wfc1 = (bf16_t*)alloc((size_t)1 * 256 * 256 * 2);
    bf16_t* wfc2 = (bf16_t*)alloc((size_t)9 * 256 * 256 * 2);
    float*  bnb  = (float*) alloc((size_t)8 * 256 * 4);
    float *sc12 = bnb,        *sh12 = bnb + 256;
    float *scp  = bnb + 512,  *shp  = bnb + 768;
    float *scc1 = bnb + 1024, *shc1 = bnb + 1280;
    float *scc2 = bnb + 1536, *shc2 = bnb + 1792;

    // dynamic LDS: 2 * (A strip + B panel), per CIN
    const int sh256 = 2 * (64 * (256 * 2 + 16) + (256 / 32) * 5120);  // 149504 B
    const int sh128 = 2 * (64 * (128 * 2 + 16) + (128 / 32) * 5120);  //  75776 B

    // 1) layout transform
    nchw_to_nhwc_bf16<<<dim3(HWSZ / 32, DC / 32, NB), dim3(32, 8), 0, stream>>>(x, xh);

    // 2) weight + BN packing
    wprep<<<(9 * 256 * 256 + 255) / 256, 256, 0, stream>>>(w_p1, w_p2, 128, 256, 256, 9, wf12);
    wprep<<<(9 * 128 * 256 + 255) / 256, 256, 0, stream>>>(w_p, w_p, 256, 128, 256, 9, wfp);
    wprep<<<(1 * 256 * 256 + 255) / 256, 256, 0, stream>>>(w_c1, w_c1, 256, 256, 256, 1, wfc1);
    wprep<<<(9 * 256 * 256 + 255) / 256, 256, 0, stream>>>(w_c2, w_c2, 256, 256, 256, 9, wfc2);
    bnprep<<<1, 256, 0, stream>>>(g_p1, b_p1, m_p1, v_p1, g_p2, b_p2, m_p2, v_p2, 128, 256, sc12, sh12);
    bnprep<<<1, 256, 0, stream>>>(g_p, b_p, m_p, v_p, g_p, b_p, m_p, v_p, 256, 256, scp, shp);
    bnprep<<<1, 256, 0, stream>>>(g_c1, b_c1, m_c1, v_c1, g_c1, b_c1, m_c1, v_c1, 256, 256, scc1, shc1);
    bnprep<<<1, 256, 0, stream>>>(g_c2, b_c2, m_c2, v_c2, g_c2, b_c2, m_c2, v_c2, 256, 256, scc2, shc2);

    const dim3 cgrid(MTOT / 64, 4);   // 1024 x 4 blocks, 4 waves each
    const int  cthreads = 128;

    // 3) fused p1|p2 conv -> f32 NHWC (256 ch), relu
    conv_wmma<256, 256, 9, true, 0, false><<<cgrid, cthreads, sh256, stream>>>(
        xh, wf12, sc12, sh12, nullptr, buf1);

    // 4) corner pools (in place) + sum -> bf16
    cummax_h<<<256, 256, 0, stream>>>(buf1);
    cummax_w<<<256, 256, 0, stream>>>(buf1);
    sum_to_bf16<<<(MTOT * 128) / 256, 256, 0, stream>>>(buf1, sbuf);

    // 5) p conv (128 -> 256), BN only -> f32 NHWC (reuse buf1)
    conv_wmma<128, 256, 9, false, 0, false><<<cgrid, cthreads, sh128, stream>>>(
        sbuf, wfp, scp, shp, nullptr, buf1);

    // 6) 1x1 residual conv + BN + add(p) + relu -> bf16 NHWC
    conv_wmma<256, 256, 1, true, 1, true><<<cgrid, cthreads, sh256, stream>>>(
        xh, wfc1, scc1, shc1, buf1, pbuf);

    // 7) final conv + BN + relu -> f32 NCHW into d_out
    conv_wmma<256, 256, 9, true, 2, false><<<cgrid, cthreads, sh256, stream>>>(
        pbuf, wfc2, scc2, shc2, nullptr, d_out);
}
